// SelfAttention_19456201851186
// MI455X (gfx1250) — matrix operand order repaired and checked by reading, not compiled, not run on hardware
//
#include <hip/hip_runtime.h>
#include <hip/hip_bf16.h>

typedef __bf16 bf16;
typedef __attribute__((ext_vector_type(16))) __bf16 v16bf;
typedef __attribute__((ext_vector_type(8)))  float  v8f;

union FragAB { v16bf v; uint4 q[2]; };   // 32 bytes: 16 bf16 halves
union PackBF8 { uint4 q; __bf16 h[8]; }; // 16 bytes

#define BB 4
#define CC 512
#define QQ 64
#define NN 4096

__device__ __forceinline__ v8f wmma_bf16(v16bf a, v16bf b, v8f c) {
  return __builtin_amdgcn_wmma_f32_16x16x32_bf16(false, a, false, b, (short)0, c, false, false);
}

// ---------------- Kernel 1: x (B,C,N) f32 -> xT (B,N,C) bf16 ----------------
__global__ __launch_bounds__(256) void k_transpose_x(const float* __restrict__ x,
                                                     bf16* __restrict__ xT) {
  __shared__ float tile[32][33];
  const int b  = blockIdx.z;
  const int n0 = blockIdx.x * 32;
  const int c0 = blockIdx.y * 32;
  const float* xb = x + (size_t)b * CC * NN;
  bf16* xTb = xT + (size_t)b * NN * CC;
  const int tx = threadIdx.x, ty = threadIdx.y;  // 32 x 8
#pragma unroll
  for (int k = 0; k < 4; ++k)
    tile[ty + 8 * k][tx] = xb[(size_t)(c0 + ty + 8 * k) * NN + n0 + tx];
  __syncthreads();
#pragma unroll
  for (int k = 0; k < 4; ++k)
    xTb[(size_t)(n0 + ty + 8 * k) * CC + c0 + tx] = (bf16)tile[tx][ty + 8 * k];
}

// ---------------- Kernel 1b: convert weights to bf16 ----------------
__global__ __launch_bounds__(256) void k_convert_w(const float* __restrict__ wq,
                                                   const float* __restrict__ wk,
                                                   const float* __restrict__ wv,
                                                   bf16* __restrict__ wqb,
                                                   bf16* __restrict__ wkb,
                                                   bf16* __restrict__ wvb) {
  const int i = blockIdx.x * blockDim.x + threadIdx.x;
  if (i < QQ * CC) { wqb[i] = (bf16)wq[i]; wkb[i] = (bf16)wk[i]; }
  if (i < CC * CC) { wvb[i] = (bf16)wv[i]; }
}

// ---------------- Kernel 2: f = wq@x, g = wk@x, h = wv@x ----------------
// fT,gT stored (B,N,64) n-major; h stored (B,C,N) c-major.
__global__ __launch_bounds__(128) void k_proj(const bf16* __restrict__ xT,
                                              const bf16* __restrict__ wqb,
                                              const bf16* __restrict__ wkb,
                                              const bf16* __restrict__ wvb,
                                              bf16* __restrict__ fT,
                                              bf16* __restrict__ gT,
                                              bf16* __restrict__ h) {
  __shared__ float hrep[4][16][16];  // per-wave transpose tile for h
  const int b    = blockIdx.y;
  const int lane = threadIdx.x & 31;
  const int w    = threadIdx.x >> 5;       // wave 0..3
  const int l16  = lane & 15;
  const int lh   = lane >> 4;              // lane-half 0/1
  const int n0   = blockIdx.x * 64 + w * 16;

  const bf16* xTb = xT + ((size_t)b * NN + n0) * CC;

  // Hold all 16 B-fragments (this wave's 16 columns, full K=512) in VGPRs.
  FragAB Bf[16];
#pragma unroll
  for (int kc = 0; kc < 16; ++kc) {
    const bf16* p = xTb + (size_t)l16 * CC + kc * 32 + lh * 16;
    Bf[kc].q[0] = *(const uint4*)(p);
    Bf[kc].q[1] = *(const uint4*)(p + 8);
  }

  for (int r = 0; r < 40; ++r) {
    const bf16* W;
    int rt;
    if (r < 4)      { W = wqb; rt = r; }
    else if (r < 8) { W = wkb; rt = r - 4; }
    else            { W = wvb; rt = r - 8; }

    const bf16* wrow = W + (size_t)(rt * 16 + l16) * CC;
    v8f acc = {};
#pragma unroll
    for (int kc = 0; kc < 16; ++kc) {
      FragAB Af;
      const bf16* p = wrow + kc * 32 + lh * 16;
      Af.q[0] = *(const uint4*)(p);
      Af.q[1] = *(const uint4*)(p + 8);
      acc = wmma_bf16(Af.v, Bf[kc].v, acc);
    }

    if (r < 8) {
      // D frag: lane col n=l16, rows m = v + 8*lh -> contiguous in fT row
      PackBF8 pk;
#pragma unroll
      for (int v2 = 0; v2 < 8; ++v2) pk.h[v2] = (bf16)acc[v2];
      bf16* dst = (r < 4 ? fT : gT) +
                  ((size_t)b * NN + n0 + l16) * QQ + rt * 16 + lh * 8;
      *(uint4*)dst = pk.q;
    } else {
      // repack 16x16 tile through LDS so h (C,N) stores are 16B contiguous
#pragma unroll
      for (int v2 = 0; v2 < 8; ++v2) hrep[w][v2 + lh * 8][l16] = acc[v2];
      __syncthreads();
      PackBF8 pk;
#pragma unroll
      for (int v2 = 0; v2 < 8; ++v2) pk.h[v2] = (bf16)hrep[w][l16][lh * 8 + v2];
      bf16* dst = h + (size_t)b * CC * NN + (size_t)(rt * 16 + l16) * NN + n0 + lh * 8;
      *(uint4*)dst = pk.q;
      __syncthreads();
    }
  }
}

// ---------------- Kernel 3: softmax stats m,l over i for each column j ----------------
__global__ __launch_bounds__(256) void k_stats(const bf16* __restrict__ fT,
                                               const bf16* __restrict__ gT,
                                               float* __restrict__ mArr,
                                               float* __restrict__ lArr) {
  const int b    = blockIdx.y;
  const int lane = threadIdx.x & 31;
  const int w    = threadIdx.x >> 5;  // 0..7
  const int l16  = lane & 15;
  const int lh   = lane >> 4;
  const int j    = blockIdx.x * 128 + w * 16 + l16;

  const bf16* fTb = fT + (size_t)b * NN * QQ;
  const bf16* gTb = gT + (size_t)b * NN * QQ;

  // B fragments (G columns j), K=64 in two chunks; loaded once.
  FragAB G0, G1;
  {
    const bf16* p = gTb + (size_t)j * QQ + lh * 16;
    G0.q[0] = *(const uint4*)(p);      G0.q[1] = *(const uint4*)(p + 8);
    G1.q[0] = *(const uint4*)(p + 32); G1.q[1] = *(const uint4*)(p + 40);
  }

  float m = -1e30f, l = 0.f;
  for (int it = 0; it < NN / 16; ++it) {
    const bf16* p = fTb + (size_t)(it * 16 + l16) * QQ + lh * 16;
    FragAB A0, A1;
    A0.q[0] = *(const uint4*)(p);      A0.q[1] = *(const uint4*)(p + 8);
    A1.q[0] = *(const uint4*)(p + 32); A1.q[1] = *(const uint4*)(p + 40);
    v8f e = {};
    e = wmma_bf16(A0.v, G0.v, e);
    e = wmma_bf16(A1.v, G1.v, e);

    float tm = e[0];
#pragma unroll
    for (int v2 = 1; v2 < 8; ++v2) tm = fmaxf(tm, e[v2]);
    tm = fmaxf(tm, __shfl_xor(tm, 16, 32));   // fuse lane-half split (rows 0-7 vs 8-15)
    const float mn = fmaxf(m, tm);
    const float sc = __expf(m - mn);
    float s = 0.f;
#pragma unroll
    for (int v2 = 0; v2 < 8; ++v2) s += __expf(e[v2] - mn);
    s += __shfl_xor(s, 16, 32);
    l = l * sc + s;
    m = mn;
  }
  if (lh == 0) {
    mArr[(size_t)b * NN + j] = m;
    lArr[(size_t)b * NN + j] = l;
  }
}

// ---------------- Kernel 4: O = H * softmax(E); out = gamma*O + x ----------------
// 8 waves; jT=64, cT=256 (2 c-subtiles per wave). Double-buffered P tile.
__global__ __launch_bounds__(256) void k_attn_out(const bf16* __restrict__ fT,
                                                  const bf16* __restrict__ gT,
                                                  const bf16* __restrict__ h,
                                                  const float* __restrict__ mArr,
                                                  const float* __restrict__ lArr,
                                                  const float* __restrict__ x,
                                                  const float* __restrict__ gamma,
                                                  float* __restrict__ out) {
  __shared__ bf16  Pl[2][64 * 32];      // double-buffered P tile: [j_local 64][i_local 32]
  __shared__ float Ot[8][2][16][64];    // output repack: [wave][c-sub][c_local][j_local]

  const int b    = blockIdx.z;
  const int c0   = blockIdx.y * 256;
  const int j0   = blockIdx.x * 64;
  const int lane = threadIdx.x & 31;
  const int w    = threadIdx.x >> 5;  // 0..7
  const int l16  = lane & 15;
  const int lh   = lane >> 4;

  const bf16* fTb = fT + (size_t)b * NN * QQ;
  const bf16* gTb = gT + (size_t)b * NN * QQ;
  const bf16* hb  = h  + (size_t)b * CC * NN;

  // Phase-A role: this wave computes E subtile (iw, jt) of the 32x64 P tile.
  const int iw = w >> 2, jt = w & 3;
  const int jA = j0 + jt * 16 + l16;
  const float mj   = mArr[(size_t)b * NN + jA];
  const float invl = 1.0f / lArr[(size_t)b * NN + jA];
  FragAB G0, G1;
  {
    const bf16* gp = gTb + (size_t)jA * QQ + lh * 16;
    G0.q[0] = *(const uint4*)(gp);      G0.q[1] = *(const uint4*)(gp + 8);
    G1.q[0] = *(const uint4*)(gp + 32); G1.q[1] = *(const uint4*)(gp + 40);
  }

  // Phase-B role: this wave owns 32 channel rows c0 + 32w .. +31.
  const bf16* hrow0 = hb + (size_t)(c0 + w * 32 + l16) * NN;
  const bf16* hrow1 = hrow0 + (size_t)16 * NN;

  v8f acc[2][4] = {{v8f{}, v8f{}, v8f{}, v8f{}}, {v8f{}, v8f{}, v8f{}, v8f{}}};

  for (int it = 0; it < NN / 32; ++it) {
    const int buf = it & 1;
    // ---- Phase A: cooperative 32x64 P = softmax tile into LDS ----
    const int i0 = it * 32 + iw * 16;
    const bf16* fp = fTb + (size_t)(i0 + l16) * QQ + lh * 16;
    FragAB A0, A1;
    A0.q[0] = *(const uint4*)(fp);      A0.q[1] = *(const uint4*)(fp + 8);
    A1.q[0] = *(const uint4*)(fp + 32); A1.q[1] = *(const uint4*)(fp + 40);
    v8f e = {};
    e = wmma_bf16(A0.v, G0.v, e);
    e = wmma_bf16(A1.v, G1.v, e);

    PackBF8 pk;
#pragma unroll
    for (int v2 = 0; v2 < 8; ++v2)
      pk.h[v2] = (bf16)(__expf(e[v2] - mj) * invl);
    *(uint4*)&Pl[buf][(jt * 16 + l16) * 32 + iw * 16 + lh * 8] = pk.q;
    __syncthreads();

    // ---- Phase B: O(32-channel slice) += H(c,i) * P(i,j), 4 j-subtiles ----
    FragAB H0, H1;
    {
      const bf16* hp0 = hrow0 + it * 32 + lh * 16;
      const bf16* hp1 = hrow1 + it * 32 + lh * 16;
      H0.q[0] = *(const uint4*)(hp0); H0.q[1] = *(const uint4*)(hp0 + 8);
      H1.q[0] = *(const uint4*)(hp1); H1.q[1] = *(const uint4*)(hp1 + 8);
    }
#pragma unroll
    for (int jt2 = 0; jt2 < 4; ++jt2) {
      FragAB P;
      const bf16* pl = &Pl[buf][(size_t)(jt2 * 16 + l16) * 32 + lh * 16];
      P.q[0] = *(const uint4*)(pl);
      P.q[1] = *(const uint4*)(pl + 8);
      acc[0][jt2] = wmma_bf16(H0.v, P.v, acc[0][jt2]);
      acc[1][jt2] = wmma_bf16(H1.v, P.v, acc[1][jt2]);
    }
    // no trailing barrier: next iteration writes the other P buffer; reuse of
    // this buffer is two barriers away.
  }

  // ---- Epilogue: out = gamma*O + x, repacked via LDS for coalesced stores ----
#pragma unroll
  for (int cs = 0; cs < 2; ++cs)
#pragma unroll
    for (int jt2 = 0; jt2 < 4; ++jt2)
#pragma unroll
      for (int v2 = 0; v2 < 8; ++v2)
        Ot[w][cs][v2 + lh * 8][jt2 * 16 + l16] = acc[cs][jt2][v2];
  __syncthreads();

  const float g0 = gamma[0];
#pragma unroll
  for (int cs = 0; cs < 2; ++cs) {
    const size_t rowoff = (size_t)b * CC * NN +
                          (size_t)(c0 + w * 32 + cs * 16 + l16) * NN + j0 + lh * 32;
    const float* xb = x + rowoff;
    float* ob = out + rowoff;
#pragma unroll
    for (int k = 0; k < 32; k += 4) {
      float4 o4 = *(const float4*)&Ot[w][cs][l16][lh * 32 + k];
      float4 x4 = *(const float4*)&xb[k];
      float4 r4;
      r4.x = g0 * o4.x + x4.x;
      r4.y = g0 * o4.y + x4.y;
      r4.z = g0 * o4.z + x4.z;
      r4.w = g0 * o4.w + x4.w;
      *(float4*)&ob[k] = r4;
    }
  }
}

// ---------------- host side ----------------
extern "C" void kernel_launch(void* const* d_in, const int* in_sizes, int n_in,
                              void* d_out, int out_size, void* d_ws, size_t ws_size,
                              hipStream_t stream) {
  (void)in_sizes; (void)n_in; (void)out_size; (void)ws_size;
  const float* x     = (const float*)d_in[0];
  const float* wq    = (const float*)d_in[1];
  const float* wk    = (const float*)d_in[2];
  const float* wv    = (const float*)d_in[3];
  const float* gamma = (const float*)d_in[4];
  float* out = (float*)d_out;
  char* ws = (char*)d_ws;

  constexpr size_t XT_B = (size_t)BB * NN * CC * 2;  // 16 MiB
  constexpr size_t WQ_B = (size_t)QQ * CC * 2;
  constexpr size_t WV_B = (size_t)CC * CC * 2;
  constexpr size_t FT_B = (size_t)BB * NN * QQ * 2;  // 2 MiB
  constexpr size_t H_B  = (size_t)BB * CC * NN * 2;  // 16 MiB
  constexpr size_t M_B  = (size_t)BB * NN * 4;

  size_t off = 0;
  auto take = [&](size_t bytes) { char* p = ws + off; off = (off + bytes + 255) & ~(size_t)255; return p; };
  bf16*  xT  = (bf16*)take(XT_B);
  bf16*  wqb = (bf16*)take(WQ_B);
  bf16*  wkb = (bf16*)take(WQ_B);
  bf16*  wvb = (bf16*)take(WV_B);
  bf16*  fTd = (bf16*)take(FT_B);
  bf16*  gTd = (bf16*)take(FT_B);
  bf16*  hd  = (bf16*)take(H_B);
  float* mA  = (float*)take(M_B);
  float* lA  = (float*)take(M_B);

  k_transpose_x<<<dim3(NN / 32, CC / 32, BB), dim3(32, 8), 0, stream>>>(x, xT);
  k_convert_w<<<(CC * CC + 255) / 256, 256, 0, stream>>>(wq, wk, wv, wqb, wkb, wvb);
  k_proj<<<dim3(NN / 64, BB), 128, 0, stream>>>(xT, wqb, wkb, wvb, fTd, gTd, hd);
  k_stats<<<dim3(NN / 128, BB), 256, 0, stream>>>(fTd, gTd, mA, lA);
  k_attn_out<<<dim3(NN / 64, CC / 256, BB), 256, 0, stream>>>(fTd, gTd, hd, mA, lA, x, gamma, out);
}
